// CausalSelfAttention_24326694764711
// MI455X (gfx1250) — matrix operand order repaired and checked by reading, not compiled, not run on hardware
//
#include <hip/hip_runtime.h>

#define DEV __device__ __forceinline__

typedef __attribute__((ext_vector_type(16))) __bf16 bf16x16;
typedef __attribute__((ext_vector_type(8)))  float f32x8;
typedef __attribute__((ext_vector_type(8)))  unsigned short u16x8;
typedef __attribute__((ext_vector_type(16))) unsigned short u16x16;
typedef __attribute__((ext_vector_type(4)))  unsigned short u16x4;

// Problem constants
static constexpr int Bn = 4;
static constexpr int Tn = 2048;
static constexpr int Cn = 768;
static constexpr int Hn = 12;
static constexpr int HD = 64;
static constexpr int NQ = Bn * Hn * Tn * HD;  // 6291456 elems (== B*T*C)

DEV unsigned short f32_to_bf16(float f) {
  __bf16 h = (__bf16)f;                       // native cvt, RNE
  return __builtin_bit_cast(unsigned short, h);
}

// ---- CDNA5 async global->LDS copy (ASYNCcnt path), 16 bytes per lane. ----
// GVS form: uniform 64-bit base in SGPRs + per-lane 32-bit byte offset.
DEV void async_ld16(const unsigned short* lds_dst, unsigned byte_off,
                    const unsigned short* base) {
  unsigned laddr = (unsigned)(size_t)lds_dst;  // addr[31:0] == LDS byte addr
  asm volatile("global_load_async_to_lds_b128 %0, %1, %2"
               :: "v"(laddr), "v"(byte_off), "s"(base) : "memory");
}
DEV void wait_async0() {
  asm volatile("s_wait_asynccnt 0x0" ::: "memory");
}

// ---- DPP16 row_xmask butterfly max over 16-lane groups (no LDS pipe). ----
template <int MASK>
DEV float dpp_xmask_f(float x) {
  int v = __builtin_amdgcn_update_dpp(0, __builtin_bit_cast(int, x),
                                      0x160 | MASK, 0xf, 0xf, true);
  return __builtin_bit_cast(float, v);
}
DEV float rowmax16(float x) {
  x = fmaxf(x, dpp_xmask_f<1>(x));
  x = fmaxf(x, dpp_xmask_f<2>(x));
  x = fmaxf(x, dpp_xmask_f<4>(x));
  x = fmaxf(x, dpp_xmask_f<8>(x));
  return x;
}

// Load a 16x32 bf16 WMMA fragment as two 16-byte LDS chunks.
DEV bf16x16 ld_frag(const unsigned short* p, int off0, int off1) {
  u16x8 lo = *(const u16x8*)(p + off0);
  u16x8 hi = *(const u16x8*)(p + off1);
  u16x16 c;
#pragma unroll
  for (int i = 0; i < 8; ++i) { c[i] = lo[i]; c[i + 8] = hi[i]; }
  return __builtin_bit_cast(bf16x16, c);
}

DEV bf16x16 ones_frag() {
  u16x16 c;
#pragma unroll
  for (int i = 0; i < 16; ++i) c[i] = 0x3F80;  // bf16 1.0
  return __builtin_bit_cast(bf16x16, c);
}

DEV f32x8 wmma_bf16(bf16x16 a, bf16x16 b, f32x8 c) {
  return __builtin_amdgcn_wmma_f32_16x16x32_bf16(false, a, false, b, (short)0, c,
                                                 false, false);
}

DEV f32x8 zero8() {
  f32x8 z;
#pragma unroll
  for (int i = 0; i < 8; ++i) z[i] = 0.0f;
  return z;
}

// ---------------------------------------------------------------------------
// Prep kernels: one-time fp32 -> bf16 conversion / transposition so every hot
// tile is a contiguous bf16 row and can be staged with async b128 copies.
// ---------------------------------------------------------------------------
__global__ __launch_bounds__(256) void cvt_rows_kernel(
    const float* __restrict__ in, unsigned short* __restrict__ out, int n4) {
  int i = blockIdx.x * 256 + threadIdx.x;
  if (i < n4) {
    const float* s = in + (size_t)i * 4;
    u16x4 o;
#pragma unroll
    for (int j = 0; j < 4; ++j) o[j] = f32_to_bf16(s[j]);
    *(u16x4*)(out + (size_t)i * 4) = o;
  }
}

// in: [K][N] fp32 -> out: [N][K] bf16   (i = n*K + k, writes coalesced)
__global__ __launch_bounds__(256) void transpose_cvt_kernel(
    const float* __restrict__ in, unsigned short* __restrict__ out, int K, int N) {
  int i = blockIdx.x * 256 + threadIdx.x;
  if (i < K * N) {
    int n = i / K, k = i - n * K;
    out[i] = f32_to_bf16(in[(size_t)k * N + n]);
  }
}

// ---------------------------------------------------------------------------
// Unified bf16 WMMA GEMM: C = A[M][768] * Bt[N][768]^T + bias, 128x128 tile,
// 8 waves (32x64 subtile each), double-buffered async LDS staging.
// EPI==0: scatter to q/k/vT bf16 buffers.  EPI==1: fp32 out.
// ---------------------------------------------------------------------------
template <int EPI>
__global__ __launch_bounds__(256) void gemm_bf16_kernel(
    const unsigned short* __restrict__ A, const unsigned short* __restrict__ Bt,
    const float* __restrict__ bias, unsigned short* __restrict__ qd,
    unsigned short* __restrict__ kd, unsigned short* __restrict__ vd,
    float* __restrict__ out) {
  __shared__ __align__(16) unsigned short As[2][128 * 40];
  __shared__ __align__(16) unsigned short Bs[2][128 * 40];

  const int tid  = threadIdx.x;
  const int wave = tid >> 5;
  const int lane = tid & 31;
  const int ln   = lane & 15;
  const int hi   = lane >> 4;
  const int wm   = (wave & 3) * 32;
  const int wn   = (wave >> 2) * 64;
  const int mbase = blockIdx.y * 128;
  const int nbase = blockIdx.x * 128;

  f32x8 acc[2][4];
#pragma unroll
  for (int mi = 0; mi < 2; ++mi)
#pragma unroll
    for (int ni = 0; ni < 4; ++ni) acc[mi][ni] = zero8();

  auto stage = [&](int buf, int k0) {
#pragma unroll
    for (int c = 0; c < 2; ++c) {  // 512 chunks per operand
      int idx = c * 256 + tid;
      int row = idx >> 2, cc = (idx & 3) * 8;
      async_ld16(&As[buf][row * 40 + cc],
                 (unsigned)(((mbase + row) * Cn + k0 + cc) * 2), A);
      async_ld16(&Bs[buf][row * 40 + cc],
                 (unsigned)(((nbase + row) * Cn + k0 + cc) * 2), Bt);
    }
  };

  stage(0, 0);
  wait_async0();
  __syncthreads();

  int buf = 0;
  for (int k0 = 0; k0 < Cn; k0 += 32) {
    if (k0 + 32 < Cn) stage(buf ^ 1, k0 + 32);  // overlap DMA with WMMA

    bf16x16 af[2], bf[4];
#pragma unroll
    for (int mi = 0; mi < 2; ++mi) {
      int row = wm + mi * 16 + ln;
      af[mi] = ld_frag(As[buf], row * 40 + hi * 8, row * 40 + 16 + hi * 8);
    }
#pragma unroll
    for (int ni = 0; ni < 4; ++ni) {
      int row = wn + ni * 16 + ln;
      bf[ni] = ld_frag(Bs[buf], row * 40 + hi * 8, row * 40 + 16 + hi * 8);
    }
#pragma unroll
    for (int mi = 0; mi < 2; ++mi)
#pragma unroll
      for (int ni = 0; ni < 4; ++ni)
        acc[mi][ni] = wmma_bf16(af[mi], bf[ni], acc[mi][ni]);

    buf ^= 1;
    wait_async0();
    __syncthreads();
  }

  if (EPI == 0) {
    // 3C splits [s][h][d]; nbase multiple of 128, 768%128==0 -> s uniform.
    const int s = nbase / Cn;
    const int colbase = nbase - s * Cn;
#pragma unroll
    for (int ni = 0; ni < 4; ++ni) {
      const int rem = colbase + wn + ni * 16 + ln;  // h*64 + d
      const int h = rem >> 6, d = rem & 63;
      const float bv = bias[nbase + wn + ni * 16 + ln];
#pragma unroll
      for (int mi = 0; mi < 2; ++mi) {
#pragma unroll
        for (int r = 0; r < 8; ++r) {
          int mg = mbase + wm + mi * 16 + r + hi * 8;
          int b = mg >> 11, t = mg & (Tn - 1);
          unsigned short val = f32_to_bf16(acc[mi][ni][r] + bv);
          if (s == 0)
            qd[((size_t)(b * Hn + h) * Tn + t) * HD + d] = val;
          else if (s == 1)
            kd[((size_t)(b * Hn + h) * Tn + t) * HD + d] = val;
          else  // V stored transposed: [bh][d][t] for async attention staging
            vd[((size_t)(b * Hn + h) * HD + d) * Tn + t] = val;
        }
      }
    }
  } else {
#pragma unroll
    for (int ni = 0; ni < 4; ++ni) {
      int ng = nbase + wn + ni * 16 + ln;
      float bv = bias[ng];
#pragma unroll
      for (int mi = 0; mi < 2; ++mi) {
#pragma unroll
        for (int r = 0; r < 8; ++r) {
          int mg = mbase + wm + mi * 16 + r + hi * 8;
          out[(size_t)mg * Cn + ng] = acc[mi][ni][r] + bv;
        }
      }
    }
  }
}

// ---------------------------------------------------------------------------
// Fused causal flash attention: one (bh, 128-query tile) per block, 8 waves,
// 64-key steps, fp32 online softmax, double-buffered async K/V staging.
// Row-max: DPP16 row_xmask butterfly.  Row-sum: extra WMMA against ones.
// ---------------------------------------------------------------------------
__global__ __launch_bounds__(256) void flash_attn_kernel(
    const unsigned short* __restrict__ q, const unsigned short* __restrict__ k,
    const unsigned short* __restrict__ vT, unsigned short* __restrict__ y) {
  __shared__ __align__(16) unsigned short Qs[128 * 72];     // [q][d]
  __shared__ __align__(16) unsigned short Ks[2][64 * 72];   // [key][d]
  __shared__ __align__(16) unsigned short Vs[2][64 * 72];   // [d][key]
  __shared__ __align__(16) unsigned short Ps[8 * 16 * 72];  // per-wave P

  const int tid  = threadIdx.x;
  const int wave = tid >> 5;
  const int lane = tid & 31;
  const int ln   = lane & 15;
  const int hi   = lane >> 4;
  const int bh    = blockIdx.y;
  const int qbase = blockIdx.x * 128;
  const int bq = bh / Hn, hq = bh % Hn;

  const unsigned short* qg  = q  + (size_t)bh * Tn * HD;
  const unsigned short* kg  = k  + (size_t)bh * Tn * HD;
  const unsigned short* vtg = vT + (size_t)bh * HD * Tn;

  auto stageKV = [&](int buf, int kb) {
#pragma unroll
    for (int c = 0; c < 2; ++c) {  // 512 chunks per operand
      int idx = c * 256 + tid;
      int r = idx >> 3, cc = (idx & 7) * 8;
      async_ld16(&Ks[buf][r * 72 + cc], (unsigned)(((kb + r) * HD + cc) * 2), kg);
      async_ld16(&Vs[buf][r * 72 + cc], (unsigned)((r * Tn + kb + cc) * 2), vtg);
    }
  };

  // Stage Q (128x64 = 1024 chunks) + first K/V tile.
#pragma unroll
  for (int c = 0; c < 4; ++c) {
    int idx = c * 256 + tid;
    int r = idx >> 3, cc = (idx & 7) * 8;
    async_ld16(&Qs[r * 72 + cc], (unsigned)(((qbase + r) * HD + cc) * 2), qg);
  }
  stageKV(0, 0);
  wait_async0();
  __syncthreads();

  bf16x16 qf[2];
  const int qrow = wave * 16 + ln;
  qf[0] = ld_frag(Qs, qrow * 72 + hi * 8,      qrow * 72 + 16 + hi * 8);
  qf[1] = ld_frag(Qs, qrow * 72 + 32 + hi * 8, qrow * 72 + 48 + hi * 8);
  const bf16x16 ones = ones_frag();

  f32x8 O[4];
#pragma unroll
  for (int j = 0; j < 4; ++j) O[j] = zero8();
  float m_run[8], l_run[8];
#pragma unroll
  for (int r = 0; r < 8; ++r) { m_run[r] = -1e30f; l_run[r] = 0.0f; }

  const int q_row_g = qbase + wave * 16 + hi * 8;
  const int nk = qbase + 128;
  int buf = 0;

  for (int kb = 0; kb < nk; kb += 64) {
    if (kb + 64 < nk) stageKV(buf ^ 1, kb + 64);  // prefetch next tile

    // S = (Q K^T) * scale, causal-masked
    f32x8 S[4];
#pragma unroll
    for (int j = 0; j < 4; ++j) {
      S[j] = zero8();
      int krow = (j * 16 + ln) * 72;
      bf16x16 b0 = ld_frag(Ks[buf], krow + hi * 8,      krow + 16 + hi * 8);
      bf16x16 b1 = ld_frag(Ks[buf], krow + 32 + hi * 8, krow + 48 + hi * 8);
      S[j] = wmma_bf16(qf[0], b0, S[j]);
      S[j] = wmma_bf16(qf[1], b1, S[j]);
    }
#pragma unroll
    for (int j = 0; j < 4; ++j) {
#pragma unroll
      for (int r = 0; r < 8; ++r) {
        float xv = S[j][r] * 0.125f;  // 1/sqrt(64)
        if (kb + j * 16 + ln > q_row_g + r) xv = -__builtin_inff();
        S[j][r] = xv;
      }
    }

    // Row max via DPP butterfly; exp; rescale O.
    float corr[8];
#pragma unroll
    for (int r = 0; r < 8; ++r) {
      float mx = fmaxf(fmaxf(S[0][r], S[1][r]), fmaxf(S[2][r], S[3][r]));
      mx = rowmax16(mx);
      float nm = fmaxf(m_run[r], mx);
      corr[r] = __expf(m_run[r] - nm);
      m_run[r] = nm;
#pragma unroll
      for (int j = 0; j < 4; ++j) S[j][r] = __expf(S[j][r] - nm);
#pragma unroll
      for (int j = 0; j < 4; ++j) O[j][r] *= corr[r];
    }

    // P (C-layout) -> LDS -> A-layout fragments (per-wave region, no barrier).
    const int pbase = wave * 16 * 72;
#pragma unroll
    for (int j = 0; j < 4; ++j)
#pragma unroll
      for (int r = 0; r < 8; ++r)
        Ps[pbase + (r + hi * 8) * 72 + j * 16 + ln] = f32_to_bf16(S[j][r]);

    bf16x16 pf0 = ld_frag(Ps, pbase + ln * 72 + hi * 8,
                              pbase + ln * 72 + 16 + hi * 8);
    bf16x16 pf1 = ld_frag(Ps, pbase + ln * 72 + 32 + hi * 8,
                              pbase + ln * 72 + 48 + hi * 8);

    // Row sums of P via WMMA against all-ones B: Z[r] = rowsum, replicated.
    f32x8 Z = zero8();
    Z = wmma_bf16(pf0, ones, Z);
    Z = wmma_bf16(pf1, ones, Z);
#pragma unroll
    for (int r = 0; r < 8; ++r) l_run[r] = l_run[r] * corr[r] + Z[r];

    // O += P V
#pragma unroll
    for (int j2 = 0; j2 < 4; ++j2) {
      int drow = (j2 * 16 + ln) * 72;
      bf16x16 v0 = ld_frag(Vs[buf], drow + hi * 8,      drow + 16 + hi * 8);
      bf16x16 v1 = ld_frag(Vs[buf], drow + 32 + hi * 8, drow + 48 + hi * 8);
      O[j2] = wmma_bf16(pf0, v0, O[j2]);
      O[j2] = wmma_bf16(pf1, v1, O[j2]);
    }

    buf ^= 1;
    wait_async0();
    __syncthreads();
  }

  // Normalize; write y[b*T + t][h*64 + d] bf16 for the projection GEMM.
#pragma unroll
  for (int j2 = 0; j2 < 4; ++j2) {
#pragma unroll
    for (int r = 0; r < 8; ++r) {
      float ov = O[j2][r] / l_run[r];
      int tg  = qbase + wave * 16 + r + hi * 8;
      int col = hq * HD + j2 * 16 + ln;
      y[((size_t)(bq * Tn + tg)) * Cn + col] = f32_to_bf16(ov);
    }
  }
}

extern "C" void kernel_launch(void* const* d_in, const int* in_sizes, int n_in,
                              void* d_out, int out_size, void* d_ws,
                              size_t ws_size, hipStream_t stream) {
  (void)in_sizes; (void)n_in; (void)out_size; (void)ws_size;
  const float* x      = (const float*)d_in[0];
  const float* W_attn = (const float*)d_in[1];
  const float* b_attn = (const float*)d_in[2];
  const float* W_proj = (const float*)d_in[3];
  const float* b_proj = (const float*)d_in[4];
  float* out = (float*)d_out;

  unsigned short* ws = (unsigned short*)d_ws;
  unsigned short* qd  = ws;                        // [B*H][T][HD] bf16
  unsigned short* kd  = ws + (size_t)NQ;           // [B*H][T][HD] bf16
  unsigned short* vtd = ws + (size_t)2 * NQ;       // [B*H][HD][T] bf16 (transposed)
  unsigned short* yd  = ws + (size_t)3 * NQ;       // [B*T][C]    bf16
  unsigned short* xb  = ws + (size_t)4 * NQ;       // [B*T][C]    bf16
  unsigned short* Wat = ws + (size_t)5 * NQ;       // [3C][C]     bf16 (W_attn^T)
  unsigned short* Wpt = Wat + (size_t)3 * Cn * Cn; // [C][C]      bf16 (W_proj^T)

  // One-time precision/layout prep (makes all hot tiles async-b128-stageable).
  cvt_rows_kernel<<<NQ / (256 * 4), 256, 0, stream>>>(x, xb, NQ / 4);
  transpose_cvt_kernel<<<(3 * Cn * Cn + 255) / 256, 256, 0, stream>>>(
      W_attn, Wat, Cn, 3 * Cn);
  transpose_cvt_kernel<<<(Cn * Cn + 255) / 256, 256, 0, stream>>>(
      W_proj, Wpt, Cn, Cn);

  // QKV projection: M=8192 (64 tiles), N=2304 (18 tiles)
  gemm_bf16_kernel<0><<<dim3(18, 64), 256, 0, stream>>>(
      xb, Wat, b_attn, qd, kd, vtd, nullptr);
  // Fused causal attention: 16 query tiles x 48 (b,h) pairs
  flash_attn_kernel<<<dim3(16, 48), 256, 0, stream>>>(qd, kd, vtd, yd);
  // Output projection: M=8192 (64 tiles), N=768 (6 tiles)
  gemm_bf16_kernel<1><<<dim3(6, 64), 256, 0, stream>>>(
      yd, Wpt, b_proj, nullptr, nullptr, nullptr, out);
}
